// RadialSelfAttention1D_89472758710669
// MI455X (gfx1250) — matrix operand order repaired and checked by reading, not compiled, not run on hardware
//
#include <hip/hip_runtime.h>
#include <hip/hip_bf16.h>

typedef __bf16 bf16;
typedef __attribute__((ext_vector_type(16))) __bf16 v16bf;
typedef __attribute__((ext_vector_type(8)))  __bf16 v8bf;
typedef __attribute__((ext_vector_type(4)))  __bf16 v4bf;
typedef __attribute__((ext_vector_type(8)))  float  v8f;
typedef __attribute__((ext_vector_type(4)))  float  v4f;
typedef __attribute__((ext_vector_type(4)))  int    v4i;

#define AS1 __attribute__((address_space(1)))
#define AS3 __attribute__((address_space(3)))

#define S_TOT  3072   // B*T
#define DMODEL 1024
#define NHEAD  16
#define HDIM   64
#define TLEN   1536

#if __has_builtin(__builtin_amdgcn_global_load_async_to_lds_b128) && __has_builtin(__builtin_amdgcn_s_wait_asynccnt)
#define HAS_ASYNC 1
#else
#define HAS_ASYNC 0
#endif

__device__ __forceinline__ v8f vzero8() {
    v8f z;
#pragma unroll
    for (int i = 0; i < 8; ++i) z[i] = 0.0f;
    return z;
}

// Copy 16 bytes global -> LDS. Async (ASYNCcnt) path when the toolchain exposes it.
__device__ __forceinline__ void stage16(const bf16* g, bf16* l) {
#if HAS_ASYNC
    __builtin_amdgcn_global_load_async_to_lds_b128(
        (AS1 v4i*)(g), (AS3 v4i*)(l), 0, 0);
#else
    *reinterpret_cast<v8bf*>(l) = *reinterpret_cast<const v8bf*>(g);
#endif
}

__device__ __forceinline__ void wait_stage() {
#if HAS_ASYNC
    __builtin_amdgcn_s_wait_asynccnt(0);
#endif
}

// A-fragment (16x32 bf16, M x K), row-major memory, `pitch` bf16 per row.
// Lane l: row l&15; K = (l>>4)*16 + {0..15}.
__device__ __forceinline__ v16bf load_a_frag(const bf16* base, int pitch, int lane) {
    const int row = lane & 15;
    const int h   = (lane >> 4) & 1;
    const bf16* p = base + row * pitch + h * 16;
    v8bf lo = *reinterpret_cast<const v8bf*>(p);
    v8bf hi = *reinterpret_cast<const v8bf*>(p + 8);
    v16bf r;
#pragma unroll
    for (int i = 0; i < 8; ++i) { r[i] = lo[i]; r[i + 8] = hi[i]; }
    return r;
}

// B-fragment (32x16 bf16, K x N). Memory holds, per column n, contiguous K values
// (`rstride` bf16 between columns). Lane n: K=0..15; lane n+16: K=16..31.
__device__ __forceinline__ v16bf load_b_frag(const bf16* base, int rstride, int lane) {
    const int n = lane & 15;
    const int h = (lane >> 4) & 1;
    const bf16* p = base + (size_t)n * rstride + h * 16;
    v8bf lo = *reinterpret_cast<const v8bf*>(p);
    v8bf hi = *reinterpret_cast<const v8bf*>(p + 8);
    v16bf r;
#pragma unroll
    for (int i = 0; i < 8; ++i) { r[i] = lo[i]; r[i + 8] = hi[i]; }
    return r;
}

__device__ __forceinline__ v8f wmma_bf16(v16bf a, v16bf b, v8f c) {
    return __builtin_amdgcn_wmma_f32_16x16x32_bf16(
        false, a, false, b, (short)0, c, false, false);
}

// ---------------------------------------------------------------------------
// Kernel 0: f32 -> bf16 pre-convert (x, Wqkv, Wout), vectorized x4.
// ---------------------------------------------------------------------------
__global__ __launch_bounds__(256) void cvt_kernel(const float* __restrict__ in,
                                                  bf16* __restrict__ out, int n4) {
    int i = blockIdx.x * 256 + threadIdx.x;
    if (i < n4) {
        v4f v = reinterpret_cast<const v4f*>(in)[i];
        v4bf b;
#pragma unroll
        for (int j = 0; j < 4; ++j) b[j] = (bf16)v[j];
        reinterpret_cast<v4bf*>(out)[i] = b;
    }
}

// Stage a ROWSxK32 bf16 tile (src row pitch `sp`) into LDS (pitch 48), 256 threads.
template <int ROWS>
__device__ __forceinline__ void stage_tile(const bf16* __restrict__ src, size_t sp,
                                           bf16* lds, int tid) {
#pragma unroll
    for (int i = 0; i < ROWS / 64; ++i) {     // ROWS*4 chunks of 8 bf16
        int idx = tid + 256 * i;
        int row = idx >> 2, c8 = idx & 3;
        stage16(src + (size_t)row * sp + c8 * 8, lds + row * 48 + c8 * 8);
    }
}

// ---------------------------------------------------------------------------
// Kernel 1: qkv = x @ Wqkv^T + bqkv  (bf16 in, f32 acc). WG tile 128x256,
// 8 waves of 64x64 (16 WMMA/k-step). Double-buffered async LDS staging.
// Q[H][S][64], K[H][S][64], VT[H][64][S] outputs; (part, head) wave-uniform.
// ---------------------------------------------------------------------------
__global__ __launch_bounds__(256) void qkv_gemm_kernel(
    const bf16* __restrict__ xb, const bf16* __restrict__ Wb,
    const float* __restrict__ bias,
    bf16* __restrict__ Qb, bf16* __restrict__ Kb, bf16* __restrict__ VTb)
{
    __shared__ __attribute__((aligned(16))) bf16 As[2][128 * 48];
    __shared__ __attribute__((aligned(16))) bf16 Bs[2][256 * 48];

    const int tid  = threadIdx.x;
    const int lane = tid & 31;
    const int w    = tid >> 5;
    const int wm   = w & 1;       // 64-row slab
    const int wn   = w >> 1;      // 64-col slab (0..3)
    const int m0   = blockIdx.y * 128;
    const int n0   = blockIdx.x * 256;

    v8f acc[4][4];
#pragma unroll
    for (int i = 0; i < 4; ++i)
#pragma unroll
        for (int j = 0; j < 4; ++j) acc[i][j] = vzero8();

    const bf16* Abase = xb + (size_t)m0 * DMODEL;
    const bf16* Bbase = Wb + (size_t)n0 * DMODEL;

    stage_tile<128>(Abase, DMODEL, As[0], tid);
    stage_tile<256>(Bbase, DMODEL, Bs[0], tid);

    int buf = 0;
    for (int kb = 0; kb < DMODEL; kb += 32) {
        wait_stage();
        __syncthreads();
        if (kb + 32 < DMODEL) {
            stage_tile<128>(Abase + kb + 32, DMODEL, As[buf ^ 1], tid);
            stage_tile<256>(Bbase + kb + 32, DMODEL, Bs[buf ^ 1], tid);
        }
        v16bf afrag[4], bfrag[4];
#pragma unroll
        for (int im = 0; im < 4; ++im)
            afrag[im] = load_a_frag(As[buf] + (wm * 64 + im * 16) * 48, 48, lane);
#pragma unroll
        for (int jn = 0; jn < 4; ++jn)
            bfrag[jn] = load_b_frag(Bs[buf] + (wn * 64 + jn * 16) * 48, 48, lane);
#pragma unroll
        for (int im = 0; im < 4; ++im)
#pragma unroll
            for (int jn = 0; jn < 4; ++jn)
                acc[im][jn] = wmma_bf16(afrag[im], bfrag[jn], acc[im][jn]);
        buf ^= 1;
    }

    // Epilogue: (part, head) are uniform across the wave's 64-aligned 64-wide slab.
    const int half = lane >> 4, col = lane & 15;
    const int n_base = n0 + wn * 64;
    const int part = n_base >> 10;          // 0=q 1=k 2=v (scalar branch)
    const int head = (n_base & 1023) >> 6;

    float bv[4];
#pragma unroll
    for (int jn = 0; jn < 4; ++jn) bv[jn] = bias[n_base + jn * 16 + col];

    if (part == 2) {
        // VT[head][d][s]: the C-fragment g-axis is contiguous in s -> packed stores.
        bf16* V = VTb + (size_t)head * HDIM * S_TOT;
#pragma unroll
        for (int im = 0; im < 4; ++im) {
            int s0 = m0 + wm * 64 + im * 16 + half * 8;
#pragma unroll
            for (int jn = 0; jn < 4; ++jn) {
                int d = jn * 16 + col;
                v8bf pk;
#pragma unroll
                for (int g = 0; g < 8; ++g) pk[g] = (bf16)(acc[im][jn][g] + bv[jn]);
                *reinterpret_cast<v8bf*>(V + (size_t)d * S_TOT + s0) = pk;
            }
        }
    } else {
        bf16* dst = (part == 0 ? Qb : Kb) + (size_t)head * S_TOT * HDIM;
#pragma unroll
        for (int im = 0; im < 4; ++im) {
#pragma unroll
            for (int jn = 0; jn < 4; ++jn) {
#pragma unroll
                for (int g = 0; g < 8; ++g) {
                    int s = m0 + wm * 64 + im * 16 + half * 8 + g;
                    dst[(size_t)s * HDIM + jn * 16 + col] = (bf16)(acc[im][jn][g] + bv[jn]);
                }
            }
        }
    }
}

// ---------------------------------------------------------------------------
// Kernel 2: fused flash attention. One (head, 128-query block) per WG,
// 4 waves x 32 rows. No workgroup barriers: each wave's P tile is private
// LDS and DS ops are in-order per wave.
// ---------------------------------------------------------------------------
__global__ __launch_bounds__(128) void attn_kernel(
    const bf16* __restrict__ Qb, const bf16* __restrict__ Kb,
    const bf16* __restrict__ VTb, bf16* __restrict__ Y)
{
    __shared__ __attribute__((aligned(16))) bf16 Plds[4][32 * 80];

    const int tid  = threadIdx.x;
    const int lane = tid & 31;
    const int w    = tid >> 5;
    const int h    = blockIdx.y;
    const int q0   = blockIdx.x * 128 + w * 32;
    const int half = lane >> 4, col = lane & 15;

    const bf16* Qh = Qb + (size_t)h * S_TOT * HDIM;
    const bf16* Kh = Kb + (size_t)h * S_TOT * HDIM;
    const bf16* Vh = VTb + (size_t)h * HDIM * S_TOT;

    v16bf qf[2][2];
#pragma unroll
    for (int m = 0; m < 2; ++m)
#pragma unroll
        for (int kk = 0; kk < 2; ++kk)
            qf[m][kk] = load_a_frag(Qh + (size_t)(q0 + m * 16) * HDIM + kk * 32, HDIM, lane);

    float mrow[2][8], lrow[2][8];
    v8f O[2][4];
#pragma unroll
    for (int m = 0; m < 2; ++m) {
#pragma unroll
        for (int g = 0; g < 8; ++g) { mrow[m][g] = -1e30f; lrow[m][g] = 0.0f; }
#pragma unroll
        for (int t = 0; t < 4; ++t) O[m][t] = vzero8();
    }

    for (int kv = 0; kv < S_TOT; kv += 64) {
        // Prefetch next KV block into cache while we compute this one.
        if (kv + 64 < S_TOT) {
            __builtin_prefetch(Kh + (size_t)(kv + 64 + lane * 2) * HDIM, 0, 3);
            __builtin_prefetch(Vh + (size_t)(lane * 2) * S_TOT + kv + 64, 0, 3);
        }

        // S = Q @ K^T (32 x 64 per wave)
        v8f sacc[2][4];
#pragma unroll
        for (int m = 0; m < 2; ++m)
#pragma unroll
            for (int j = 0; j < 4; ++j) sacc[m][j] = vzero8();

#pragma unroll
        for (int j = 0; j < 4; ++j) {
#pragma unroll
            for (int kk = 0; kk < 2; ++kk) {
                v16bf kf = load_b_frag(Kh + (size_t)(kv + j * 16) * HDIM + kk * 32, HDIM, lane);
#pragma unroll
                for (int m = 0; m < 2; ++m)
                    sacc[m][j] = wmma_bf16(qf[m][kk], kf, sacc[m][j]);
            }
        }

        // Online softmax (rows replicated over 16-lane halves)
#pragma unroll
        for (int m = 0; m < 2; ++m) {
#pragma unroll
            for (int g = 0; g < 8; ++g) {
                float v = -1e30f;
#pragma unroll
                for (int j = 0; j < 4; ++j) {
                    sacc[m][j][g] *= 0.125f;   // 1/sqrt(64)
                    v = fmaxf(v, sacc[m][j][g]);
                }
#pragma unroll
                for (int msk = 1; msk < 16; msk <<= 1)
                    v = fmaxf(v, __shfl_xor(v, msk, 32));
                float mo   = mrow[m][g];
                float mn   = fmaxf(mo, v);
                float corr = __expf(mo - mn);
                mrow[m][g] = mn;
                float rs = 0.0f;
#pragma unroll
                for (int j = 0; j < 4; ++j) {
                    float p = __expf(sacc[m][j][g] - mn);
                    sacc[m][j][g] = p;
                    rs += p;
                }
#pragma unroll
                for (int msk = 1; msk < 16; msk <<= 1)
                    rs += __shfl_xor(rs, msk, 32);
                lrow[m][g] = lrow[m][g] * corr + rs;
#pragma unroll
                for (int t = 0; t < 4; ++t) O[m][t][g] *= corr;
            }
        }

        // C-layout -> A-layout via this wave's private LDS tile. LDS ops are
        // in-order per wave, so no workgroup barrier is needed.
        bf16* P = &Plds[w][0];
#pragma unroll
        for (int m = 0; m < 2; ++m)
#pragma unroll
            for (int j = 0; j < 4; ++j)
#pragma unroll
                for (int g = 0; g < 8; ++g)
                    P[(m * 16 + half * 8 + g) * 80 + j * 16 + col] = (bf16)sacc[m][j][g];
        __builtin_amdgcn_wave_barrier();

        v16bf pf[2][2];
#pragma unroll
        for (int m = 0; m < 2; ++m)
#pragma unroll
            for (int kk = 0; kk < 2; ++kk)
                pf[m][kk] = load_a_frag(P + m * 16 * 80 + kk * 32, 80, lane);

        // O += P @ V (V pre-transposed: contiguous B-fragments)
#pragma unroll
        for (int t = 0; t < 4; ++t) {
#pragma unroll
            for (int kk = 0; kk < 2; ++kk) {
                v16bf vf = load_b_frag(Vh + (size_t)(t * 16) * S_TOT + kv + kk * 32, S_TOT, lane);
#pragma unroll
                for (int m = 0; m < 2; ++m)
                    O[m][t] = wmma_bf16(pf[m][kk], vf, O[m][t]);
            }
        }
        __builtin_amdgcn_wave_barrier();
    }

    // Normalize, apply the reference's (T,B) row shuffle, store bf16.
#pragma unroll
    for (int m = 0; m < 2; ++m) {
#pragma unroll
        for (int g = 0; g < 8; ++g) {
            int s = q0 + m * 16 + half * 8 + g;
            int r = (s & 1) * TLEN + (s >> 1);
            float inv = 1.0f / lrow[m][g];
#pragma unroll
            for (int t = 0; t < 4; ++t) {
                int d = t * 16 + col;
                Y[(size_t)r * DMODEL + h * HDIM + d] = (bf16)(O[m][t][g] * inv);
            }
        }
    }
}

// ---------------------------------------------------------------------------
// Kernel 3: out = y @ Wout^T + bout  (M=3072, N=1024, K=1024), f32 out.
// WG tile 128x128, 8 waves of 64x32, double-buffered async staging.
// ---------------------------------------------------------------------------
__global__ __launch_bounds__(256) void out_gemm_kernel(
    const bf16* __restrict__ Yb, const bf16* __restrict__ Wb,
    const float* __restrict__ bias, float* __restrict__ out)
{
    __shared__ __attribute__((aligned(16))) bf16 As[2][128 * 48];
    __shared__ __attribute__((aligned(16))) bf16 Bs[2][128 * 48];

    const int tid  = threadIdx.x;
    const int lane = tid & 31;
    const int w    = tid >> 5;
    const int wm   = w & 1;
    const int wn   = w >> 1;
    const int m0   = blockIdx.y * 128;
    const int n0   = blockIdx.x * 128;

    v8f acc[4][2];
#pragma unroll
    for (int i = 0; i < 4; ++i)
#pragma unroll
        for (int j = 0; j < 2; ++j) acc[i][j] = vzero8();

    const bf16* Abase = Yb + (size_t)m0 * DMODEL;
    const bf16* Bbase = Wb + (size_t)n0 * DMODEL;

    stage_tile<128>(Abase, DMODEL, As[0], tid);
    stage_tile<128>(Bbase, DMODEL, Bs[0], tid);

    int buf = 0;
    for (int kb = 0; kb < DMODEL; kb += 32) {
        wait_stage();
        __syncthreads();
        if (kb + 32 < DMODEL) {
            stage_tile<128>(Abase + kb + 32, DMODEL, As[buf ^ 1], tid);
            stage_tile<128>(Bbase + kb + 32, DMODEL, Bs[buf ^ 1], tid);
        }
        v16bf afrag[4], bfrag[2];
#pragma unroll
        for (int im = 0; im < 4; ++im)
            afrag[im] = load_a_frag(As[buf] + (wm * 64 + im * 16) * 48, 48, lane);
#pragma unroll
        for (int jn = 0; jn < 2; ++jn)
            bfrag[jn] = load_b_frag(Bs[buf] + (wn * 32 + jn * 16) * 48, 48, lane);
#pragma unroll
        for (int im = 0; im < 4; ++im)
#pragma unroll
            for (int jn = 0; jn < 2; ++jn)
                acc[im][jn] = wmma_bf16(afrag[im], bfrag[jn], acc[im][jn]);
        buf ^= 1;
    }

    const int half = lane >> 4, col = lane & 15;
#pragma unroll
    for (int im = 0; im < 4; ++im) {
#pragma unroll
        for (int jn = 0; jn < 2; ++jn) {
#pragma unroll
            for (int g = 0; g < 8; ++g) {
                int s = m0 + wm * 64 + im * 16 + half * 8 + g;
                int n = n0 + wn * 32 + jn * 16 + col;
                out[(size_t)s * DMODEL + n] = acc[im][jn][g] + bias[n];
            }
        }
    }
}

extern "C" void kernel_launch(void* const* d_in, const int* in_sizes, int n_in,
                              void* d_out, int out_size, void* d_ws, size_t ws_size,
                              hipStream_t stream) {
    const float* x    = (const float*)d_in[0];
    const float* Wqkv = (const float*)d_in[1];
    const float* bqkv = (const float*)d_in[2];
    const float* Wout = (const float*)d_in[3];
    const float* bout = (const float*)d_in[4];

    const size_t nXD = (size_t)S_TOT * DMODEL;           // 3.1M
    const size_t nW  = (size_t)DMODEL * DMODEL;          // 1.0M
    bf16* xb    = (bf16*)d_ws;
    bf16* Wqkvb = xb + nXD;            // 3*D x D == S_TOT x D here (3072x1024)
    bf16* Woutb = Wqkvb + nXD;
    bf16* Qb    = Woutb + nW;
    bf16* Kb    = Qb + nXD;
    bf16* VTb   = Kb + nXD;
    bf16* Yb    = VTb + nXD;

    // 0) precision pre-pass: f32 -> bf16 once
    cvt_kernel<<<dim3((unsigned)(nXD / 4 / 256)), 256, 0, stream>>>(x, xb, (int)(nXD / 4));
    cvt_kernel<<<dim3((unsigned)(nXD / 4 / 256)), 256, 0, stream>>>(Wqkv, Wqkvb, (int)(nXD / 4));
    cvt_kernel<<<dim3((unsigned)(nW / 4 / 256)), 256, 0, stream>>>(Wout, Woutb, (int)(nW / 4));

    // 1) QKV projection: grid (N/256, M/128) = (12, 24)
    qkv_gemm_kernel<<<dim3(12, 24), 256, 0, stream>>>(xb, Wqkvb, bqkv, Qb, Kb, VTb);
    // 2) Fused attention: grid (S/128, H) = (24, 16)
    attn_kernel<<<dim3(24, 16), 128, 0, stream>>>(Qb, Kb, VTb, Yb);
    // 3) Output projection: grid (N/128, M/128) = (8, 24)
    out_gemm_kernel<<<dim3(8, 24), 256, 0, stream>>>(Yb, Woutb, bout, (float*)d_out);
}
